// ContextIterator_66726611911121
// MI455X (gfx1250) — matrix operand order repaired and checked by reading, not compiled, not run on hardware
//
#include <hip/hip_runtime.h>

typedef __attribute__((ext_vector_type(16))) _Float16 v16h;
typedef __attribute__((ext_vector_type(8)))  _Float16 v8h;
typedef __attribute__((ext_vector_type(8)))  float    v8f;
typedef __attribute__((ext_vector_type(4)))  unsigned int u32x4;
typedef __attribute__((ext_vector_type(8)))  int          i32x8;
typedef __attribute__((ext_vector_type(4)))  int          i32x4;

#define BTOT 8
#define HTOT 64
#define WTOT 64
#define CTOT 320                       // total channels in y
#define NPIX (BTOT * HTOT * WTOT)      // 32768
#define XF16_ELEMS ((size_t)NPIX * CTOT)  // 10,485,760 per version

// ---------------------------------------------------------------------------
// Prep 1: transpose y [B,320,64,64] f32 -> channel-fastest f16 [B,64,64,320],
// one unmasked copy and one pre-multiplied by m0 (checkerboard (h+w) even).
// ---------------------------------------------------------------------------
__global__ __launch_bounds__(256) void prep_x(const float* __restrict__ x,
                                              _Float16* __restrict__ xz,   // unmasked
                                              _Float16* __restrict__ xm)   // * m0
{
    const int idx = blockIdx.x * 256 + threadIdx.x;       // over B*320*64*64
    const int ww  = idx & 63;
    const int hh  = (idx >> 6) & 63;
    const int c   = (idx >> 12) % CTOT;
    const int b   = idx / (CTOT * 4096);
    const float v = x[idx];
    const size_t dst = ((size_t)((b * HTOT + hh) * WTOT + ww)) * CTOT + c;
    xz[dst] = (_Float16)v;
    xm[dst] = (_Float16)(v * (float)(1 - ((hh + ww) & 1)));
}

// ---------------------------------------------------------------------------
// Prep 2 (per conv): pack weights [Cout,CIN,5,5] f32 into f16 tiles laid out
// [coutBlk][slab][kh][kw][m(32)][c(SLAB)], channels zero-padded to CINPAD.
// Each (coutBlk,slab,kh) chunk of 5*32*SLAB f16 is contiguous -> TDM-friendly.
// ---------------------------------------------------------------------------
template<int CIN>
__global__ __launch_bounds__(256) void prep_w(const float* __restrict__ wgt,
                                              _Float16* __restrict__ dst,
                                              int total)
{
    constexpr int CINPAD = (CIN < 32) ? 32 : CIN;
    constexpr int SLAB   = (CINPAD < 64) ? CINPAD : 64;
    constexpr int NSLAB  = CINPAD / SLAB;

    const int idx = blockIdx.x * 256 + threadIdx.x;
    if (idx >= total) return;
    const int c  = idx % SLAB;
    const int v2 = (idx / SLAB) % (5 * 32);
    const int m  = v2 % 32;
    const int kw = v2 / 32;
    const int q  = idx / (5 * 32 * SLAB);
    const int kh = q % 5;
    const int r  = q / 5;
    const int slab    = r % NSLAB;
    const int coutBlk = r / NSLAB;
    const int cin = slab * SLAB + c;
    float v = 0.0f;
    if (cin < CIN)
        v = wgt[((coutBlk * 32 + m) * CIN + cin) * 25 + kh * 5 + kw];
    dst[idx] = (_Float16)v;
}

// ---------------------------------------------------------------------------
// Conv kernel: tiled implicit GEMM, f16 WMMA + f32 accumulate.
// Block tile: 32 couts x 64 pixels (one (b,h) row); 8 waves of 16x16 tiles.
// Input tile staged from pre-transposed f16 tensor with v8h copies; weight
// tile staged with the Tensor Data Mover (contiguous chunk -> LDS).
// ---------------------------------------------------------------------------
template<int CIN, int MASK>
__global__ __launch_bounds__(256) void conv5_wmma(
    const _Float16* __restrict__ xf16,   // [B,64,64,320] f16 (masked for sp)
    const _Float16* __restrict__ wf16,   // packed tiles (prep_w layout)
    const float*    __restrict__ bias,
    float*          __restrict__ out,    // [B, Cout, 64, 64]
    int chanOff, int Cout)
{
    constexpr int CINPAD = (CIN < 32) ? 32 : CIN;
    constexpr int SLAB   = (CINPAD < 64) ? CINPAD : 64;
    constexpr int NSLAB  = CINPAD / SLAB;
    constexpr int KCH    = SLAB / 32;
    constexpr int PPL    = SLAB / 8;            // 16B parts per (hh,ww) line
    constexpr int WCHUNK = 5 * 32 * SLAB;       // f16 per weight tile

    __shared__ _Float16 xlds[5 * 68 * SLAB];    // [hh][ww+halo][c]
    __shared__ _Float16 alds[WCHUNK];           // [kw][m][c] for current kh

    const int tid  = threadIdx.x;
    const int lane = tid & 31;
    const int wv   = tid >> 5;
    const int mi   = wv & 1;
    const int ni   = wv >> 1;
    const int half = lane >> 4;
    const int nlo  = lane & 15;

    const int rowIdx   = blockIdx.x & 511;      // b*64 + h
    const int coutBlk  = blockIdx.x >> 9;
    const int b        = rowIdx >> 6;
    const int h        = rowIdx & 63;
    const int coutBase = coutBlk * 32;

    v8f acc = {};

    for (int slab = 0; slab < NSLAB; ++slab) {
        const int cin0 = slab * SLAB;
        __syncthreads();   // previous slab's xlds reads complete

        // ---- stage input halo tile: clamped always-issued 16B loads ----
        for (int idx = tid; idx < 340 * PPL; idx += 256) {
            const int line = idx / PPL;
            const int p    = idx - line * PPL;
            const int hh   = line / 68;
            const int ww   = line - hh * 68;
            const int gh   = h + hh - 2;
            const int gw   = ww - 2;
            const int ghc  = gh < 0 ? 0 : (gh > 63 ? 63 : gh);
            const int gwc  = gw < 0 ? 0 : (gw > 63 ? 63 : gw);
            const int cb   = cin0 + p * 8;
            v8h v = *(const v8h*)(xf16 +
                     ((size_t)((b * HTOT + ghc) * WTOT + gwc)) * CTOT + chanOff + cb);
            const bool ok = ((unsigned)gh < 64u) && ((unsigned)gw < 64u) && (cb < CIN);
            if (!ok) v = (v8h){};
            *(v8h*)(&xlds[line * SLAB + p * 8]) = v;
        }

        for (int kh = 0; kh < 5; ++kh) {
            __syncthreads();   // alds consumers done / xlds staged
            const _Float16* wsrc =
                wf16 + ((size_t)(coutBlk * NSLAB + slab) * 5 + kh) * WCHUNK;

#if __has_builtin(__builtin_amdgcn_tensor_load_to_lds)
            if (wv == 0) {
                // 1-D TDM descriptor: copy WCHUNK f16 (contiguous) -> alds
                const unsigned int N  = (unsigned int)WCHUNK;       // elements
                const unsigned int lo = (unsigned int)(uintptr_t)(void*)alds;
                const unsigned long long ga = (unsigned long long)(uintptr_t)wsrc;
                u32x4 g0 = { 1u,                                  // count=1
                             lo,                                  // lds_addr
                             (unsigned int)(ga & 0xFFFFFFFFull),
                             (unsigned int)((ga >> 32) & 0x01FFFFFFull) | (2u << 30) };
                i32x8 g1 = { (int)(1u << 16),                     // data_size=2B
                             (int)((N & 0xFFFFu) << 16),          // tensor_dim0 lo
                             (int)(((N >> 16) & 0xFFFFu) | (1u << 16)), // dim0 hi | dim1=1
                             (int)(N << 16),                      // tile_dim0
                             1,                                   // tile_dim1=1
                             (int)N,                              // dim0_stride
                             0, 0 };
                i32x4 z4 = { 0, 0, 0, 0 };
#if defined(__clang_major__) && (__clang_major__ >= 23)
                i32x8 z8 = { 0, 0, 0, 0, 0, 0, 0, 0 };
                __builtin_amdgcn_tensor_load_to_lds(g0, g1, z4, z4, z8, 0);
#else
                __builtin_amdgcn_tensor_load_to_lds(g0, g1, z4, z4, 0);
#endif
                __builtin_amdgcn_s_wait_tensorcnt(0);
            }
#else
            for (int idx = tid; idx < WCHUNK / 8; idx += 256)
                *(v8h*)(&alds[idx * 8]) = *(const v8h*)(wsrc + idx * 8);
#endif
            __syncthreads();   // weight tile visible to all waves

            // ---- hot loop: pure LDS reads + WMMA, fully unrolled ----
            #pragma unroll
            for (int kw = 0; kw < 5; ++kw) {
                #pragma unroll
                for (int ch = 0; ch < KCH; ++ch) {
                    const int cb = ch * 32;
                    const _Float16* ap = &alds[(kw * 32 + mi * 16 + nlo) * SLAB + cb];
                    v8h a0 = *(const v8h*)(ap + half * 16);
                    v8h a1 = *(const v8h*)(ap + half * 16 + 8);
                    v16h af;
                    #pragma unroll
                    for (int e = 0; e < 8; ++e) { af[e] = a0[e]; af[8 + e] = a1[e]; }
                    const _Float16* bp =
                        &xlds[(kh * 68 + ni * 16 + nlo + kw) * SLAB + cb + half * 16];
                    v16h bf = *(const v16h*)bp;
                    acc = __builtin_amdgcn_wmma_f32_16x16x32_f16(
                        false, af, false, bf, (short)0, acc, false, false);
                }
            }
        }
    }

    const int   wn      = ni * 16 + nlo;
    const float outMask = MASK ? (float)((h + wn) & 1) : 1.0f;   // * m1
    #pragma unroll
    for (int r = 0; r < 8; ++r) {
        const int m = coutBase + mi * 16 + r + half * 8;
        out[((b * Cout + m) * HTOT + h) * WTOT + wn] = (acc[r] + bias[m]) * outMask;
    }
}

// ---------------------------------------------------------------------------
template<int CIN, int MASK>
static void run_conv(const _Float16* xf16, const float* wgt, _Float16* wpak,
                     const float* bias, float* out, int chanOff, int Cout,
                     hipStream_t stream)
{
    constexpr int CINPAD = (CIN < 32) ? 32 : CIN;
    const int wtotal = Cout * CINPAD * 25;
    prep_w<CIN><<<dim3((unsigned)((wtotal + 255) / 256)), 256, 0, stream>>>(wgt, wpak, wtotal);
    dim3 grid((unsigned)((Cout / 32) * (BTOT * HTOT)));
    conv5_wmma<CIN, MASK><<<grid, 256, 0, stream>>>(xf16, wpak, bias, out, chanOff, Cout);
}

extern "C" void kernel_launch(void* const* d_in, const int* in_sizes, int n_in,
                              void* d_out, int out_size, void* d_ws, size_t ws_size,
                              hipStream_t stream) {
    (void)in_sizes; (void)n_in; (void)out_size; (void)ws_size;

    const float* y = (const float*)d_in[0];
    const float* sw[5] = { (const float*)d_in[3], (const float*)d_in[5],
                           (const float*)d_in[7], (const float*)d_in[9],
                           (const float*)d_in[11] };
    const float* sb[5] = { (const float*)d_in[4], (const float*)d_in[6],
                           (const float*)d_in[8], (const float*)d_in[10],
                           (const float*)d_in[12] };
    const float* cw[4] = { (const float*)d_in[13], (const float*)d_in[15],
                           (const float*)d_in[17], (const float*)d_in[19] };
    const float* cb[4] = { (const float*)d_in[14], (const float*)d_in[16],
                           (const float*)d_in[18], (const float*)d_in[20] };

    float* out = (float*)d_out;
    const size_t PLANE = (size_t)NPIX;

    // workspace layout: xz (unmasked f16) | xm (masked f16) | packed weights
    _Float16* xz = (_Float16*)d_ws;
    _Float16* xm = xz + XF16_ELEMS;
    _Float16* wp = xm + XF16_ELEMS;

    prep_x<<<dim3((unsigned)(XF16_ELEMS / 256)), 256, 0, stream>>>(y, xz, xm);

    size_t ofs = 0; _Float16* w = wp;
    // sp0: Cin=16 @0, Cout=32 (masked in/out)
    run_conv<16, 1>(xm, sw[0], w, sb[0], out + ofs, 0, 32, stream);
    ofs += 32 * PLANE; w += 32 * 32 * 25;
    // sp1: Cin=16 @16, Cout=32
    run_conv<16, 1>(xm, sw[1], w, sb[1], out + ofs, 16, 32, stream);
    ofs += 32 * PLANE; w += 32 * 32 * 25;
    // ch1: Cin=16 @0, Cout=32 (unmasked)
    run_conv<16, 0>(xz, cw[0], w, cb[0], out + ofs, 0, 32, stream);
    ofs += 32 * PLANE; w += 32 * 32 * 25;
    // sp2: Cin=32 @32, Cout=64
    run_conv<32, 1>(xm, sw[2], w, sb[2], out + ofs, 32, 64, stream);
    ofs += 64 * PLANE; w += 64 * 32 * 25;
    // ch2: Cin=32 @0, Cout=64
    run_conv<32, 0>(xz, cw[1], w, cb[1], out + ofs, 0, 64, stream);
    ofs += 64 * PLANE; w += 64 * 32 * 25;
    // sp3: Cin=64 @64, Cout=128
    run_conv<64, 1>(xm, sw[3], w, sb[3], out + ofs, 64, 128, stream);
    ofs += 128 * PLANE; w += 128 * 64 * 25;
    // ch3: Cin=64 @0, Cout=128
    run_conv<64, 0>(xz, cw[2], w, cb[2], out + ofs, 0, 128, stream);
    ofs += 128 * PLANE; w += 128 * 64 * 25;
    // sp4: Cin=192 @128, Cout=384
    run_conv<192, 1>(xm, sw[4], w, sb[4], out + ofs, 128, 384, stream);
    ofs += 384 * PLANE; w += 384 * 192 * 25;
    // ch4: Cin=128 @0, Cout=384
    run_conv<128, 0>(xz, cw[3], w, cb[3], out + ofs, 0, 384, stream);
}
